// GCNModel_22582938043087
// MI455X (gfx1250) — matrix operand order repaired and checked by reading, not compiled, not run on hardware
//
#include <hip/hip_runtime.h>
#include <hip/hip_bf16.h>
#include <stdint.h>

// ---------------------------------------------------------------------------
// 2-layer GCN for MI455X (gfx1250, wave32).
//  - Dense X@W via v_wmma_f32_16x16x32_bf16; LDS staged so fragments load as
//    ds_load_b128 (A row-major, W pre-transposed to [N][K] in bf16 so B tiles
//    stage as straight b128 copies). Rows padded to 40 u16 for bank spread.
//  - Degree / rsqrt-norm / edge scatter via L2-resident f32 atomics (float4)
//  - bias + ReLU + hash dropout fused epilogue
// ---------------------------------------------------------------------------

typedef __attribute__((ext_vector_type(16))) __bf16          v16bf;
typedef __attribute__((ext_vector_type(16))) unsigned short  u16x16;
typedef __attribute__((ext_vector_type(8)))  unsigned short  u16x8;
typedef __attribute__((ext_vector_type(8)))  float           v8f;
typedef __attribute__((ext_vector_type(4)))  float           v4f;

#define GCN_N_NODES 10000
#define GCN_N_EDGES 160000
#define GCN_IN_C    256
#define GCN_HID_C   512
#define GCN_OUT_C   256

#define LDS_PITCH   40   // u16 per row (80B): 16B-aligned chunks, row starts
                         // walk 16 distinct bank groups (m*20 mod 64)

__device__ __forceinline__ unsigned short f32_to_bf16_rne(float f) {
  unsigned int u = __builtin_bit_cast(unsigned int, f);
  u += 0x7FFFu + ((u >> 16) & 1u);           // round-to-nearest-even
  return (unsigned short)(u >> 16);
}

__device__ __forceinline__ unsigned int hash_u32(unsigned int x) {
  x ^= x >> 16; x *= 0x7feb352du;
  x ^= x >> 15; x *= 0x846ca68bu;
  x ^= x >> 16;
  return x;
}

__device__ __forceinline__ u16x16 cat16(u16x8 lo, u16x8 hi) {
  return __builtin_shufflevector(lo, hi, 0, 1, 2, 3, 4, 5, 6, 7,
                                         8, 9, 10, 11, 12, 13, 14, 15);
}

// -------- weight conversion + transpose: W[K][N] f32 -> Wbt[N][K] bf16 -----
__global__ void k_cvt_bf16_t(const float* __restrict__ W,
                             unsigned short* __restrict__ Wbt,
                             int K, int N) {
  int i = blockIdx.x * blockDim.x + threadIdx.x;
  if (i >= K * N) return;
  int k = i / N, n = i - k * N;
  Wbt[(size_t)n * K + k] = f32_to_bf16_rne(W[i]);
}

// ---------------- degree / normalization -----------------------------------
__global__ void k_deg_init(float* __restrict__ deg, int n) {
  int i = blockIdx.x * blockDim.x + threadIdx.x;
  if (i < n) deg[i] = 1.0f;                  // self-loop
}

__global__ void k_deg_accum(const int* __restrict__ dst_idx,
                            float* __restrict__ deg, int e) {
  int i = blockIdx.x * blockDim.x + threadIdx.x;
  if (i < e) atomicAdd(&deg[dst_idx[i]], 1.0f);
}

__global__ void k_dinv(const float* __restrict__ deg,
                       float* __restrict__ dinv, int n) {
  int i = blockIdx.x * blockDim.x + threadIdx.x;
  if (i < n) dinv[i] = rsqrtf(deg[i]);
}

// ---------------- bf16 WMMA GEMM: Y[M,N] = X[M,K] * Wbt[N,K]^T -------------
// 256 threads = 8 waves. Block tile 64(M) x 64(N); wave (wm 0..3, wn 0..1)
// computes a 16x32 D strip = two 16x16 WMMA tiles sharing the A fragment.
__global__ __launch_bounds__(256)
void k_gemm_bf16_wmma(const float* __restrict__ X,
                      const unsigned short* __restrict__ Wbt,  // [N][K] bf16
                      float* __restrict__ Y, int M, int K, int N) {
  __shared__ unsigned short lds_a[64 * LDS_PITCH];   // [m][k], k 0..31
  __shared__ unsigned short lds_bt[64 * LDS_PITCH];  // [n][k], k 0..31

  const int tid    = threadIdx.x;
  const int lane   = tid & 31;
  const int wave   = tid >> 5;
  const int wm     = wave & 3;          // 0..3 along M
  const int wn     = wave >> 2;         // 0..1 along N (each owns 32 cols)
  const int blockM = blockIdx.y * 64;
  const int blockN = blockIdx.x * 64;

  // staging coordinates (hoisted). Tail rows are CLAMPED, not masked:
  // A row m only feeds D row m, and D rows >= M are never stored.
  const int arow = tid >> 2;            // 0..63
  const int acol = (tid & 3) * 8;       // 0,8,16,24
  const int gm   = min(blockM + arow, M - 1);
  const int brow = tid >> 2;            // 0..63 (n row of B^T tile)
  const int bchk = (tid & 3) * 8;       // 0,8,16,24 (k chunk)
  const unsigned short* wsrc = Wbt + (size_t)(blockN + brow) * K + bchk;

  v8f acc0 = {};
  v8f acc1 = {};

  for (int k0 = 0; k0 < K; k0 += 32) {
    // ---- stage A: 2x global_load_b128 -> pack bf16 -> ds_store_b128 ------
    {
      const v4f* xp = (const v4f*)(X + (size_t)gm * K + (k0 + acol));
      v4f x0 = xp[0];
      v4f x1 = xp[1];
      u16x8 p;
#pragma unroll
      for (int j = 0; j < 4; ++j) {
        p[j]     = f32_to_bf16_rne(x0[j]);
        p[4 + j] = f32_to_bf16_rne(x1[j]);
      }
      *(u16x8*)(lds_a + arow * LDS_PITCH + acol) = p;
    }
    // ---- stage B^T: straight 16B copy (global_load_b128 + ds_store_b128) -
    {
      u16x8 w = *(const u16x8*)(wsrc + k0);
      *(u16x8*)(lds_bt + brow * LDS_PITCH + bchk) = w;
    }
    // hint next A k-slab into cache (global_prefetch_b8)
    if (k0 + 32 < K)
      __builtin_prefetch(&X[(size_t)gm * K + (k0 + 32 + acol)], 0, 1);
    __syncthreads();

    // ---- fragments: two ds_load_b128 each (ISA 7.12.2 16-bit layout) -----
    const int mn  = lane & 15;          // M row (A) / N col (B) within tile
    const int sel = lane >> 4;          // K-half select

    const unsigned short* ar = lds_a + (wm * 16 + mn) * LDS_PITCH;
    u16x16 av = cat16(*(const u16x8*)(ar + sel * 8),
                      *(const u16x8*)(ar + 16 + sel * 8));

    const unsigned short* br0 = lds_bt + (wn * 32 + mn) * LDS_PITCH + sel * 8;
    u16x16 bv0 = cat16(*(const u16x8*)(br0), *(const u16x8*)(br0 + 16));
    const unsigned short* br1 = br0 + 16 * LDS_PITCH;
    u16x16 bv1 = cat16(*(const u16x8*)(br1), *(const u16x8*)(br1 + 16));

    acc0 = __builtin_amdgcn_wmma_f32_16x16x32_bf16(
        false, __builtin_bit_cast(v16bf, av),
        false, __builtin_bit_cast(v16bf, bv0),
        (short)0, acc0, false, false);
    acc1 = __builtin_amdgcn_wmma_f32_16x16x32_bf16(
        false, __builtin_bit_cast(v16bf, av),
        false, __builtin_bit_cast(v16bf, bv1),
        (short)0, acc1, false, false);
    __syncthreads();
  }

  // D layout: VGPR r -> M = r + 8*(lane>=16); N = lane&15
  const int mn    = lane & 15;
  const int sel   = lane >> 4;
  const int n0    = blockN + wn * 32 + mn;
  const int mBase = blockM + wm * 16 + sel * 8;
#pragma unroll
  for (int r = 0; r < 8; ++r) {
    int m = mBase + r;
    if (m < M) {
      Y[(size_t)m * N + n0]      = acc0[r];
      Y[(size_t)m * N + n0 + 16] = acc1[r];
    }
  }
}

// ---------------- aggregation (float4 vectorized) --------------------------
// self-loop message initializes the output: out = xw * dinv^2
__global__ void k_agg_self(const v4f* __restrict__ xw,
                           const float* __restrict__ dinv,
                           v4f* __restrict__ out, int n, int C4) {
  long long i = (long long)blockIdx.x * blockDim.x + threadIdx.x;
  if (i >= (long long)n * C4) return;
  int node = (int)(i / C4);
  float d  = dinv[node];
  out[i] = xw[i] * (d * d);
}

// per-edge scatter: out[dst] += xw[src] * dinv[src]*dinv[dst]  (L2 atomics)
__global__ void k_agg_edges(const v4f* __restrict__ xw,
                            const float* __restrict__ dinv,
                            const int* __restrict__ srcI,
                            const int* __restrict__ dstI,
                            float* __restrict__ out, int E, int C4) {
  long long i = (long long)blockIdx.x * blockDim.x + threadIdx.x;
  if (i >= (long long)E * C4) return;
  int e  = (int)(i / C4);
  int c4 = (int)(i - (long long)e * C4);
  int s = srcI[e], d = dstI[e];
  float norm = dinv[s] * dinv[d];
  v4f v = xw[(size_t)s * C4 + c4] * norm;
  float* o = out + (size_t)d * (C4 * 4) + c4 * 4;
#pragma unroll
  for (int j = 0; j < 4; ++j) atomicAdd(o + j, v[j]);
}

// ---------------- bias + ReLU + dropout(p=0.5) epilogue --------------------
__global__ void k_post(v4f* __restrict__ y, const v4f* __restrict__ bias,
                       int n, int C4, unsigned int seed) {
  long long i = (long long)blockIdx.x * blockDim.x + threadIdx.x;
  if (i >= (long long)n * C4) return;
  int c4 = (int)(i % C4);
  v4f v = y[i] + bias[c4];
  unsigned int base = (unsigned int)(i * 4) ^ seed;
#pragma unroll
  for (int j = 0; j < 4; ++j) {
    float r = v[j] > 0.0f ? v[j] : 0.0f;
    v[j] = (hash_u32(base + j) & 0x80000000u) ? r * 2.0f : 0.0f;
  }
  y[i] = v;
}

// ---------------------------------------------------------------------------
extern "C" void kernel_launch(void* const* d_in, const int* in_sizes, int n_in,
                              void* d_out, int out_size, void* d_ws, size_t ws_size,
                              hipStream_t stream) {
  (void)in_sizes; (void)n_in; (void)out_size; (void)ws_size;

  const float* X  = (const float*)d_in[0];      // [10000, 256]
  const int*   ei = (const int*)d_in[1];        // [2, 160000]
  const float* W1 = (const float*)d_in[2];      // [256, 512]
  const float* b1 = (const float*)d_in[3];      // [512]
  const float* W2 = (const float*)d_in[4];      // [512, 256]
  const float* b2 = (const float*)d_in[5];      // [256]
  float* out = (float*)d_out;                   // [10000, 256]

  char*  ws  = (char*)d_ws;
  size_t off = 0;
  auto alloc = [&](size_t bytes) -> char* {
    char* p = ws + off;
    off += (bytes + 255) & ~(size_t)255;
    return p;
  };
  unsigned short* W1bt = (unsigned short*)alloc((size_t)GCN_IN_C  * GCN_HID_C * 2); // [HID][IN]
  unsigned short* W2bt = (unsigned short*)alloc((size_t)GCN_HID_C * GCN_OUT_C * 2); // [OUT][HID]
  float* deg  = (float*)alloc((size_t)GCN_N_NODES * 4);
  float* dinv = (float*)alloc((size_t)GCN_N_NODES * 4);
  float* bufA = (float*)alloc((size_t)GCN_N_NODES * GCN_HID_C * 4);  // xw
  float* bufB = (float*)alloc((size_t)GCN_N_NODES * GCN_HID_C * 4);  // agg/h

  const int* srcIdx = ei;
  const int* dstIdx = ei + GCN_N_EDGES;

  const int T = 256;

  // weights -> bf16, transposed to [N][K]
  k_cvt_bf16_t<<<(GCN_IN_C * GCN_HID_C + T - 1) / T, T, 0, stream>>>(W1, W1bt, GCN_IN_C, GCN_HID_C);
  k_cvt_bf16_t<<<(GCN_HID_C * GCN_OUT_C + T - 1) / T, T, 0, stream>>>(W2, W2bt, GCN_HID_C, GCN_OUT_C);

  // degrees + rsqrt
  k_deg_init<<<(GCN_N_NODES + T - 1) / T, T, 0, stream>>>(deg, GCN_N_NODES);
  k_deg_accum<<<(GCN_N_EDGES + T - 1) / T, T, 0, stream>>>(dstIdx, deg, GCN_N_EDGES);
  k_dinv<<<(GCN_N_NODES + T - 1) / T, T, 0, stream>>>(deg, dinv, GCN_N_NODES);

  // ---- layer 1: xw = X@W1 ; agg ; relu+bias+dropout ----
  {
    dim3 g(GCN_HID_C / 64, (GCN_N_NODES + 63) / 64);
    k_gemm_bf16_wmma<<<g, T, 0, stream>>>(X, W1bt, bufA, GCN_N_NODES, GCN_IN_C, GCN_HID_C);
    const int C4 = GCN_HID_C / 4;
    long long tot  = (long long)GCN_N_NODES * C4;
    long long tedg = (long long)GCN_N_EDGES * C4;
    k_agg_self<<<(unsigned)((tot + T - 1) / T), T, 0, stream>>>((const v4f*)bufA, dinv, (v4f*)bufB, GCN_N_NODES, C4);
    k_agg_edges<<<(unsigned)((tedg + T - 1) / T), T, 0, stream>>>((const v4f*)bufA, dinv, srcIdx, dstIdx, bufB, GCN_N_EDGES, C4);
    k_post<<<(unsigned)((tot + T - 1) / T), T, 0, stream>>>((v4f*)bufB, (const v4f*)b1, GCN_N_NODES, C4, 0x9E3779B9u);
  }

  // ---- layer 2: xw = h1@W2 ; agg -> d_out ; relu+bias+dropout ----
  {
    dim3 g(GCN_OUT_C / 64, (GCN_N_NODES + 63) / 64);
    k_gemm_bf16_wmma<<<g, T, 0, stream>>>(bufB, W2bt, bufA, GCN_N_NODES, GCN_HID_C, GCN_OUT_C);
    const int C4 = GCN_OUT_C / 4;
    long long tot  = (long long)GCN_N_NODES * C4;
    long long tedg = (long long)GCN_N_EDGES * C4;
    k_agg_self<<<(unsigned)((tot + T - 1) / T), T, 0, stream>>>((const v4f*)bufA, dinv, (v4f*)out, GCN_N_NODES, C4);
    k_agg_edges<<<(unsigned)((tedg + T - 1) / T), T, 0, stream>>>((const v4f*)bufA, dinv, srcIdx, dstIdx, out, GCN_N_EDGES, C4);
    k_post<<<(unsigned)((tot + T - 1) / T), T, 0, stream>>>((v4f*)out, (const v4f*)b2, GCN_N_NODES, C4, 0x85EBCA6Bu);
  }
}
